// MultiHeadAttention_9251359556148
// MI455X (gfx1250) — compile-verified, matrix operand order checked by reading, not run on hardware
//
#include <hip/hip_runtime.h>
#include <hip/hip_bf16.h>
#include <math.h>

// ---------------------------------------------------------------------------
// MHA forward for MI455X (gfx1250): bf16 WMMA everywhere, fp32 accumulate.
// B=8 S=2048 E=512 H=8 D=64.  ~103 GFLOP vs ~140MB HBM traffic -> matrix-op
// bound; all GEMMs (QKV proj, QK^T, PV, out proj) run on v_wmma_f32_16x16x32_bf16.
// K/V tiles for attention are staged block-wide into LDS with the CDNA5 async
// load-to-LDS path (ASYNCcnt), double buffered to overlap DMA with WMMA.
// ---------------------------------------------------------------------------

constexpr int Bb = 8, Ss = 2048, Ee = 512, Hh = 8, Dd = 64;
constexpr int Mm = Bb * Ss;            // 16384 rows

typedef __bf16 bf16;
typedef bf16  v16bf __attribute__((ext_vector_type(16)));
typedef bf16  v8bf  __attribute__((ext_vector_type(8)));
typedef float v8f   __attribute__((ext_vector_type(8)));
typedef float v4f   __attribute__((ext_vector_type(4)));

// fp32 -> bf16: native convert (RNE)
__device__ inline bf16 f2bf(float f) { return (bf16)f; }

__device__ inline v8f wmma_bf16(v16bf a, v16bf b, v8f c) {
  // D = A(16x32 bf16) x B(32x16 bf16) + C(16x16 f32)
  return __builtin_amdgcn_wmma_f32_16x16x32_bf16(
      /*neg_a=*/false, a, /*neg_b=*/false, b,
      /*c_mod=*/(short)0, c, /*reuse_a=*/false, /*reuse_b=*/false);
}

// 16x32 (rows x K) bf16 fragment from row-major storage with leading dim `ld`.
// CDNA5 16-bit A layout: lane l<16 holds row l, K in [k0,k0+8)+[k0+16,k0+24);
// lane l>=16 holds row l-16, K in [k0+8,k0+16)+[k0+24,k0+32).  Two b128 loads.
// Same pattern serves as B-fragment when the operand is stored N x K row-major.
__device__ inline v16bf load_frag_bf16(const bf16* p, int ld, int row0, int k0, int lane) {
  const bf16* base = p + (size_t)(row0 + (lane & 15)) * ld + k0 + ((lane >> 4) << 3);
  union { v16bf f; struct { v8bf lo; v8bf hi; } c; } u;
  u.c.lo = *(const v8bf*)base;
  u.c.hi = *(const v8bf*)(base + 16);
  return u.f;
}

// Same fragment, but source is fp32 (on-the-fly conversion, saves a full pass).
__device__ inline v16bf load_frag_f32(const float* p, int ld, int row0, int k0, int lane) {
  const float* base = p + (size_t)(row0 + (lane & 15)) * ld + k0 + ((lane >> 4) << 3);
  v4f a0 = *(const v4f*)base;
  v4f a1 = *(const v4f*)(base + 4);
  v4f b0 = *(const v4f*)(base + 16);
  v4f b1 = *(const v4f*)(base + 20);
  v16bf f;
#pragma unroll
  for (int i = 0; i < 4; ++i) {
    f[i]      = f2bf(a0[i]);
    f[i + 4]  = f2bf(a1[i]);
    f[i + 8]  = f2bf(b0[i]);
    f[i + 12] = f2bf(b1[i]);
  }
  return f;
}

// CDNA5 async DMA: 16B global -> LDS, tracked by ASYNCcnt (not LOADcnt).
__device__ inline void async_b128_to_lds(unsigned lds_byte_addr, const bf16* gptr) {
  asm volatile("global_load_async_to_lds_b128 %0, %1, off"
               :: "v"(lds_byte_addr), "v"(gptr)
               : "memory");
}
__device__ inline void wait_asynccnt0() {
  asm volatile("s_wait_asynccnt 0" ::: "memory");
}
// Generic (flat) shared pointer -> wave-relative LDS byte address (low 32 bits).
__device__ inline unsigned lds_addr(const void* p) {
  return (unsigned)(uintptr_t)p;
}

// ---------------------------------------------------------------------------
// Weight conversion fp32 -> bf16 (2MB total; trivially bandwidth bound)
// ---------------------------------------------------------------------------
__global__ void cvt_f32_to_bf16_kernel(const float* __restrict__ src,
                                       bf16* __restrict__ dst, int n) {
  int i = (blockIdx.x * blockDim.x + threadIdx.x) * 4;
  if (i + 3 < n) {
    v4f v = *(const v4f*)(src + i);
#pragma unroll
    for (int j = 0; j < 4; ++j) dst[i + j] = f2bf(v[j]);
  }
}

// ---------------------------------------------------------------------------
// GEMM: out = X @ W^T + bias.  X: [M,512] (fp32 or bf16), W: [512,512] bf16
// row-major (N x K, so B-fragments are K-contiguous).  Block = 8 waves (2x4),
// block tile 128x256, wave tile 64x64 (4x4 accumulator tiles -> 16 WMMAs per
// 16 b128 loads per K-step), 16 K-steps, unrolled x2 for software pipelining.
// OUT_MODE 0: bf16 -> [B,H,S,D]   (Q/K head-split)
// OUT_MODE 1: bf16 -> [B,H,D,S]   (V transposed for K-contiguous PV fragments)
// OUT_MODE 2: fp32 -> [M,512]     (final output)
// ---------------------------------------------------------------------------
template <int OUT_MODE, bool A_F32>
__global__ __launch_bounds__(256) void gemm_xwt_kernel(
    const void* __restrict__ Xv, const bf16* __restrict__ W,
    const float* __restrict__ bias, void* __restrict__ Outv) {
  const int lane = threadIdx.x & 31;
  const int wave = threadIdx.x >> 5;
  const int m0 = blockIdx.x * 128 + (wave & 1) * 64;
  const int n0 = blockIdx.y * 256 + (wave >> 1) * 64;

  const float* Xf = (const float*)Xv;
  const bf16*  Xb = (const bf16*)Xv;

  v8f acc[4][4] = {};

#pragma unroll 2
  for (int k0 = 0; k0 < Ee; k0 += 32) {
    if (k0 + 32 < Ee)  // keep next weight K-slab warm: global_prefetch_b8
      __builtin_prefetch((const void*)(W + (size_t)(n0 + (lane & 15)) * Ee + k0 + 32), 0, 3);
    v16bf a[4];
#pragma unroll
    for (int mi = 0; mi < 4; ++mi) {
      if constexpr (A_F32) a[mi] = load_frag_f32(Xf, Ee, m0 + mi * 16, k0, lane);
      else                 a[mi] = load_frag_bf16(Xb, Ee, m0 + mi * 16, k0, lane);
    }
#pragma unroll
    for (int t = 0; t < 4; ++t) {
      v16bf bf = load_frag_bf16(W, Ee, n0 + t * 16, k0, lane);
#pragma unroll
      for (int mi = 0; mi < 4; ++mi)
        acc[mi][t] = wmma_bf16(a[mi], bf, acc[mi][t]);
    }
  }

  // Epilogue. C layout: element r of the v8f = row (r + 8*(lane>=16)),
  // column = lane & 15.
  const int col = lane & 15;
  const int hi8 = (lane >> 4) * 8;
#pragma unroll
  for (int t = 0; t < 4; ++t) {
    const int n = n0 + t * 16 + col;
    const float bv = bias[n];
#pragma unroll
    for (int mi = 0; mi < 4; ++mi) {
#pragma unroll
      for (int r = 0; r < 8; ++r) {
        const int m = m0 + mi * 16 + r + hi8;
        const float val = acc[mi][t][r] + bv;
        if constexpr (OUT_MODE == 2) {
          ((float*)Outv)[(size_t)m * Ee + n] = val;
        } else {
          const int b = m >> 11, s = m & (Ss - 1);
          const int h = n >> 6, d = n & (Dd - 1);
          size_t idx;
          if constexpr (OUT_MODE == 0)
            idx = ((size_t)(b * Hh + h) * Ss + s) * Dd + d;
          else
            idx = ((size_t)(b * Hh + h) * Dd + d) * Ss + s;
          ((bf16*)Outv)[idx] = f2bf(val);
        }
      }
    }
  }
}

// ---------------------------------------------------------------------------
// Flash attention: grid (S/128, B*H), block 256 (8 waves). Each wave owns a
// 16-row query tile; all 8 waves share each 32-key K/V tile, which is staged
// into LDS once per block via global_load_async_to_lds_b128 (double buffered,
// DMA overlapped with WMMA + softmax of the previous tile).  Online-softmax
// stats and the 16x64 fp32 output accumulator stay in registers.  P is
// round-tripped through a per-wave-private LDS slice to convert C-layout ->
// A-fragment layout (LDS ops are in-order per wave; s_wait_dscnt orders the
// store->load pair).
// ---------------------------------------------------------------------------
__global__ __launch_bounds__(256) void flash_attn_kernel(
    const bf16* __restrict__ Qh, const bf16* __restrict__ Kh,
    const bf16* __restrict__ Vt, bf16* __restrict__ Oc) {
  __shared__ __align__(16) bf16 kbuf[2][32 * 64];   // [j 32][d 64] per buffer
  __shared__ __align__(16) bf16 vbuf[2][64 * 32];   // [d 64][j 32] per buffer
  __shared__ __align__(16) bf16 lds_p[8][16 * 32];  // per-wave P round-trip

  const int tid  = threadIdx.x;
  const int lane = tid & 31;
  const int wave = tid >> 5;
  const int bh = blockIdx.y;            // b*H + h
  const int b = bh >> 3, h = bh & 7;
  const int q0 = blockIdx.x * 128 + wave * 16;
  const int col = lane & 15;
  const int hi8 = (lane >> 4) * 8;

  const bf16* Qp = Qh + (size_t)bh * Ss * Dd;   // [S,D] row-major
  const bf16* Kp = Kh + (size_t)bh * Ss * Dd;   // [S,D] row-major
  const bf16* Vp = Vt + (size_t)bh * Dd * Ss;   // [D,S] row-major (transposed)

  // Per-thread async fill of one K chunk + one V chunk (256 x 16B each tile).
  const int krow = tid >> 3, kch = tid & 7;     // K tile: 32 rows x 8 chunks
  const int vrow = tid >> 2, vch = tid & 3;     // V tile: 64 rows x 4 chunks
  auto fill_kv = [&](int buf, int j0) {
    async_b128_to_lds(lds_addr(&kbuf[buf][0]) + tid * 16,
                      Kp + (size_t)(j0 + krow) * Dd + kch * 8);
    async_b128_to_lds(lds_addr(&vbuf[buf][0]) + tid * 16,
                      Vp + (size_t)vrow * Ss + j0 + vch * 8);
  };

  const v16bf qf0 = load_frag_bf16(Qp, Dd, q0, 0, lane);
  const v16bf qf1 = load_frag_bf16(Qp, Dd, q0, 32, lane);

  v8f acc[4] = {};
  float mrow[8], lrow[8];
#pragma unroll
  for (int r = 0; r < 8; ++r) { mrow[r] = -__builtin_inff(); lrow[r] = 0.f; }

  constexpr float scale = 0.04419417382415922f;   // 1/sqrt(E)
  constexpr float log2e = 1.4426950408889634f;
  constexpr int NJB = Ss / 32;                    // 64 key blocks

  fill_kv(0, 0);

#pragma unroll 2   // makes the buffer parity a compile-time constant
  for (int jb = 0; jb < NJB; ++jb) {
    const int cur = jb & 1;
    // Fill of buf[cur] (issued last iteration) must have landed, and all waves
    // must be past their use of the buffer we are about to refill (they last
    // touched it in iteration jb-1, before this barrier).
    wait_asynccnt0();
    __syncthreads();
    if (jb + 1 < NJB) fill_kv(1 - cur, (jb + 1) * 32);

    const bf16* kt = kbuf[cur];
    const bf16* vt = vbuf[cur];

    // ---- scores for key tiles [0,16) and [16,32) of this block (K = D = 64)
    v8f st0 = {}, st1 = {};
    st0 = wmma_bf16(qf0, load_frag_bf16(kt, Dd, 0, 0, lane), st0);
    st0 = wmma_bf16(qf1, load_frag_bf16(kt, Dd, 0, 32, lane), st0);
    st1 = wmma_bf16(qf0, load_frag_bf16(kt, Dd, 16, 0, lane), st1);
    st1 = wmma_bf16(qf1, load_frag_bf16(kt, Dd, 16, 32, lane), st1);

    // ---- online softmax over the 32 new columns
    float pm[8];
#pragma unroll
    for (int r = 0; r < 8; ++r) {
      st0[r] *= scale; st1[r] *= scale;
      pm[r] = fmaxf(st0[r], st1[r]);
    }
#pragma unroll
    for (int xm = 1; xm < 16; xm <<= 1)
#pragma unroll
      for (int r = 0; r < 8; ++r)
        pm[r] = fmaxf(pm[r], __shfl_xor(pm[r], xm, 32));

    float p0[8], p1[8], rs[8];
#pragma unroll
    for (int r = 0; r < 8; ++r) {
      const float mn = fmaxf(mrow[r], pm[r]);
      const float corr = __builtin_exp2f((mrow[r] - mn) * log2e);
      mrow[r] = mn;
      p0[r] = __builtin_exp2f((st0[r] - mn) * log2e);
      p1[r] = __builtin_exp2f((st1[r] - mn) * log2e);
      rs[r] = p0[r] + p1[r];
      lrow[r] *= corr;
#pragma unroll
      for (int t = 0; t < 4; ++t) acc[t][r] *= corr;
    }
#pragma unroll
    for (int xm = 1; xm < 16; xm <<= 1)
#pragma unroll
      for (int r = 0; r < 8; ++r)
        rs[r] += __shfl_xor(rs[r], xm, 32);
#pragma unroll
    for (int r = 0; r < 8; ++r) lrow[r] += rs[r];

    // ---- C-layout P -> row-major 16x32 in private LDS slice -> A fragment
    bf16* pb = lds_p[wave];
#pragma unroll
    for (int r = 0; r < 8; ++r) {
      const int row = r + hi8;
      pb[row * 32 + col]      = f2bf(p0[r]);
      pb[row * 32 + 16 + col] = f2bf(p1[r]);
    }
    asm volatile("s_wait_dscnt 0" ::: "memory");
    const v16bf pf = load_frag_bf16(pb, 32, 0, 0, lane);

    // ---- acc += P(16x32) @ V(32x64); LDS V tile rows are d, K-contiguous in j
#pragma unroll
    for (int t = 0; t < 4; ++t)
      acc[t] = wmma_bf16(pf, load_frag_bf16(vt, 32, t * 16, 0, lane), acc[t]);
  }

  // ---- normalize and write concat layout [B,S,E] bf16
#pragma unroll
  for (int r = 0; r < 8; ++r) {
    const float inv = 1.0f / lrow[r];
    const int s = q0 + r + hi8;
#pragma unroll
    for (int t = 0; t < 4; ++t) {
      const size_t idx = ((size_t)b * Ss + s) * Ee + h * 64 + t * 16 + col;
      Oc[idx] = f2bf(acc[t][r] * inv);
    }
  }
}

// ---------------------------------------------------------------------------
extern "C" void kernel_launch(void* const* d_in, const int* in_sizes, int n_in,
                              void* d_out, int out_size, void* d_ws, size_t ws_size,
                              hipStream_t stream) {
  const float* q  = (const float*)d_in[0];
  const float* k  = (const float*)d_in[1];
  const float* v  = (const float*)d_in[2];
  const float* Wq = (const float*)d_in[3];
  const float* bq = (const float*)d_in[4];
  const float* Wk = (const float*)d_in[5];
  const float* bk = (const float*)d_in[6];
  const float* Wv = (const float*)d_in[7];
  const float* bv = (const float*)d_in[8];
  const float* Wo = (const float*)d_in[9];
  const float* bo = (const float*)d_in[10];

  constexpr size_t WB = (size_t)Ee * Ee * sizeof(bf16);        // 512 KB
  constexpr size_t AB = (size_t)Bb * Ss * Ee * sizeof(bf16);   // 16 MB
  char* ws = (char*)d_ws;
  bf16* Wqb = (bf16*)(ws + 0 * WB);
  bf16* Wkb = (bf16*)(ws + 1 * WB);
  bf16* Wvb = (bf16*)(ws + 2 * WB);
  bf16* Wob = (bf16*)(ws + 3 * WB);
  bf16* Qh  = (bf16*)(ws + 4 * WB + 0 * AB);   // [B,H,S,D]
  bf16* Kh  = (bf16*)(ws + 4 * WB + 1 * AB);   // [B,H,S,D]
  bf16* Vt  = (bf16*)(ws + 4 * WB + 2 * AB);   // [B,H,D,S]
  bf16* Oc  = (bf16*)(ws + 4 * WB + 3 * AB);   // [B,S,E]

  // 1) weights -> bf16
  const int wn = Ee * Ee;
  cvt_f32_to_bf16_kernel<<<wn / (4 * 256), 256, 0, stream>>>(Wq, Wqb, wn);
  cvt_f32_to_bf16_kernel<<<wn / (4 * 256), 256, 0, stream>>>(Wk, Wkb, wn);
  cvt_f32_to_bf16_kernel<<<wn / (4 * 256), 256, 0, stream>>>(Wv, Wvb, wn);
  cvt_f32_to_bf16_kernel<<<wn / (4 * 256), 256, 0, stream>>>(Wo, Wob, wn);

  // 2) QKV projections (fp32 A converted in the loader)
  dim3 gproj(Mm / 128, Ee / 256);
  gemm_xwt_kernel<0, true><<<gproj, 256, 0, stream>>>((const void*)q, Wqb, bq, (void*)Qh);
  gemm_xwt_kernel<0, true><<<gproj, 256, 0, stream>>>((const void*)k, Wkb, bk, (void*)Kh);
  gemm_xwt_kernel<1, true><<<gproj, 256, 0, stream>>>((const void*)v, Wvb, bv, (void*)Vt);

  // 3) attention
  flash_attn_kernel<<<dim3(Ss / 128, Bb * Hh), 256, 0, stream>>>(Qh, Kh, Vt, Oc);

  // 4) output projection -> fp32 d_out
  gemm_xwt_kernel<2, false><<<gproj, 256, 0, stream>>>((const void*)Oc, Wob, bo, d_out);
}
